// OptimizedHilbertAttention_58926951301481
// MI455X (gfx1250) — matrix operand order repaired and checked by reading, not compiled, not run on hardware
//
#include <hip/hip_runtime.h>
#include <hip/hip_bf16.h>

typedef __attribute__((ext_vector_type(16))) __bf16    v16bf;
typedef __attribute__((ext_vector_type(8)))  float     v8f;
typedef __attribute__((ext_vector_type(4)))  unsigned  v4u;
typedef __attribute__((ext_vector_type(8)))  unsigned  v8u;

#define WMMA_BF16(A_, B_, C_) \
  __builtin_amdgcn_wmma_f32_16x16x32_bf16(false, (A_), false, (B_), (short)0, (C_), false, false)

__device__ inline unsigned pack_bf16(float a, float b) {
  union { __bf16 h[2]; unsigned u; } p;
  p.h[0] = (__bf16)a; p.h[1] = (__bf16)b;
  return p.u;   // lowers to v_cvt_pk_bf16_f32
}

// ---------------------------------------------------------------------------
// WMMA fragment loaders from LDS (CDNA5 §7.12.2 layouts, wave32).
// A 16x32 bf16: lane<16 row M=lane, K {0..7}{16..23}; lane>=16 K {8..15}{24..31}.
// B 32x16 bf16 as [n][k]: lane<16 N=lane K {0..7}{16..23}; lane>=16 N=lane-16 K {8..15}{24..31}.
// All element pairs contiguous -> b32/b128 DS loads.
// ---------------------------------------------------------------------------
__device__ inline v16bf load_a_frag(const __bf16* base, int stride, int row0, int k0, int lane) {
  const int half = (lane >> 4) & 1, r = lane & 15;
  union { v16bf v; unsigned u[8]; } f;
  const __bf16* p = base + (row0 + r) * stride + k0 + half * 8;
#pragma unroll
  for (int i = 0; i < 4; ++i) f.u[i]     = *(const unsigned*)(p + 2 * i);
#pragma unroll
  for (int i = 0; i < 4; ++i) f.u[4 + i] = *(const unsigned*)(p + 16 + 2 * i);
  return f.v;
}

__device__ inline v16bf load_b_frag(const __bf16* base, int stride, int col0, int k0, int lane) {
  const int half = (lane >> 4) & 1, c = lane & 15;
  union { v16bf v; unsigned u[8]; } f;
  const __bf16* p = base + (col0 + c) * stride + k0 + half * 8;
#pragma unroll
  for (int i = 0; i < 4; ++i) f.u[i]     = *(const unsigned*)(p + 2 * i);
#pragma unroll
  for (int i = 0; i < 4; ++i) f.u[4 + i] = *(const unsigned*)(p + 16 + 2 * i);
  return f.v;
}

// ---------------------------------------------------------------------------
// Kernel 1: fused QKV GEMM. x[8192,1024] f32 x Wqkv[1024,3072] f32 + bqkv ->
// q/k/v bf16 in [B=2][H=16][M=4096][D=64]; q pre-scaled by D^-1/2.
// 128x128 block tile, 8 waves (2x4), each wave 64x32 (8 accs), K slab 64.
// ---------------------------------------------------------------------------
__global__ __launch_bounds__(256) void qkv_gemm_kernel(
    const float* __restrict__ x, const float* __restrict__ Wqkv,
    const float* __restrict__ bqkv,
    __bf16* __restrict__ qo, __bf16* __restrict__ ko, __bf16* __restrict__ vo) {
  __shared__ __bf16 As[128 * 64];   // [r][k]
  __shared__ __bf16 Bs[128 * 64];   // [n][k]

  const int t = threadIdx.x, lane = t & 31, wave = t >> 5;
  const int wm = wave & 1, wn = wave >> 1;
  const int m0 = blockIdx.y * 128, n0 = blockIdx.x * 128;

  v8f acc[4][2] = {};

  for (int k0 = 0; k0 < 1024; k0 += 64) {
#pragma unroll
    for (int i = 0; i < 16; ++i) {      // A: 128x64, float2 loads, packed b32 stores
      int p = i * 256 + t, r = p >> 5, k2 = (p & 31) * 2;
      const float* gp = &x[(m0 + r) * 1024 + k0 + k2];
      *(unsigned*)&As[r * 64 + k2] = pack_bf16(gp[0], gp[1]);
    }
#pragma unroll
    for (int i = 0; i < 16; ++i) {      // B: [n][k] transposed stage, packed on k
      int p = i * 256 + t, n = p & 127, k2 = (p >> 7) * 2;
      const float* gp = &Wqkv[(k0 + k2) * 3072 + n0 + n];
      *(unsigned*)&Bs[n * 64 + k2] = pack_bf16(gp[0], gp[3072]);
    }
    __syncthreads();
#pragma unroll
    for (int ks = 0; ks < 64; ks += 32) {
      v16bf a[4], b[2];
#pragma unroll
      for (int im = 0; im < 4; ++im) a[im] = load_a_frag(As, 64, wm * 64 + im * 16, ks, lane);
#pragma unroll
      for (int in_ = 0; in_ < 2; ++in_) b[in_] = load_b_frag(Bs, 64, wn * 32 + in_ * 16, ks, lane);
#pragma unroll
      for (int im = 0; im < 4; ++im)
#pragma unroll
        for (int in_ = 0; in_ < 2; ++in_)
          acc[im][in_] = WMMA_BF16(a[im], b[in_], acc[im][in_]);
    }
    __syncthreads();
  }

  const int half = lane >> 4, c = lane & 15;
#pragma unroll
  for (int im = 0; im < 4; ++im)
#pragma unroll
    for (int in_ = 0; in_ < 2; ++in_) {
      int col = n0 + wn * 32 + in_ * 16 + c;            // 0..3071
      int which = col >> 10, rem = col & 1023;
      int h = rem >> 6, d = rem & 63;
      float bias  = bqkv[col];
      float scale = (which == 0) ? 0.125f : 1.0f;       // fold 64^-0.5 into q
      __bf16* dst = (which == 0) ? qo : (which == 1) ? ko : vo;
#pragma unroll
      for (int rr = 0; rr < 8; ++rr) {
        int row = m0 + wm * 64 + im * 16 + half * 8 + rr;
        int bb = row >> 12, mm = row & 4095;
        dst[(((bb * 16 + h) * 4096) + mm) * 64 + d] =
            (__bf16)((acc[im][in_][rr] + bias) * scale);
      }
    }
}

// ---------------------------------------------------------------------------
// Kernel 2: Hilbert block attention, one workgroup per (b,h,block).
// Q pre-scaled; masks dropped (M = 16*SEG exactly). QK^T and PV on WMMA.
// ---------------------------------------------------------------------------
__global__ __launch_bounds__(256) void hilbert_attn_kernel(
    const __bf16* __restrict__ q, const __bf16* __restrict__ k,
    const __bf16* __restrict__ v, const int* __restrict__ mapping,
    __bf16* __restrict__ attn) {
  __shared__ __bf16 Qs[64 * 64];   // [m][d] (pre-scaled)
  __shared__ __bf16 Ks[64 * 64];   // [s][d] == B-layout [n][k] for QK^T
  __shared__ __bf16 Vt[64 * 64];   // [d][s] == B-layout [n][k] for PV
  __shared__ __bf16 Ps[64 * 64];   // exp(S-max): A matrix of PV
  __shared__ float  Sf[64 * 64];
  __shared__ float  lsum[64];
  __shared__ int    qrow[64], krow[64];

  const int t = threadIdx.x, lane = t & 31, wave = t >> 5;
  const int wm = wave & 3, wn = wave >> 2;
  const int bid = blockIdx.x;
  const int bb = bid >> 10, hh = (bid >> 6) & 15, nb = bid & 63;

  if (t < 64) qrow[t] = mapping[nb * 64 + t];
  __syncthreads();
  const int seg = qrow[0] >> 8;                       // SEG = 256
  if (t < 64) krow[t] = mapping[seg * 256 + 4 * t];   // DIL = 4
  __syncthreads();

  const unsigned* qg = (const unsigned*)(q + (size_t)((bb * 16 + hh) * 4096) * 64);
  const unsigned* kg = (const unsigned*)(k + (size_t)((bb * 16 + hh) * 4096) * 64);
  const unsigned* vg = (const unsigned*)(v + (size_t)((bb * 16 + hh) * 4096) * 64);
  unsigned* Qs32 = (unsigned*)Qs;
  unsigned* Ks32 = (unsigned*)Ks;

#pragma unroll
  for (int i = 0; i < 8; ++i) {        // 64 rows x 32 dwords, u32 copies
    int p = i * 256 + t, m = p >> 5, j = p & 31;
    Qs32[m * 32 + j] = qg[qrow[m] * 32 + j];
    Ks32[m * 32 + j] = kg[krow[m] * 32 + j];
    union { unsigned u; __bf16 h[2]; } w;             // V transposed for B layout
    w.u = vg[krow[m] * 32 + j];
    Vt[(2 * j) * 64 + m]     = w.h[0];
    Vt[(2 * j + 1) * 64 + m] = w.h[1];
  }
  __syncthreads();

  // ---- S = Qs . K^T : 64x64, K-dim = 64 ----
  v8f sacc[2] = {};
#pragma unroll
  for (int ks = 0; ks < 64; ks += 32) {
    v16bf a  = load_a_frag(Qs, 64, wm * 16, ks, lane);
    v16bf b0 = load_b_frag(Ks, 64, wn * 32, ks, lane);
    v16bf b1 = load_b_frag(Ks, 64, wn * 32 + 16, ks, lane);
    sacc[0] = WMMA_BF16(a, b0, sacc[0]);
    sacc[1] = WMMA_BF16(a, b1, sacc[1]);
  }
  const int half = lane >> 4, c = lane & 15;
#pragma unroll
  for (int nt = 0; nt < 2; ++nt) {
    int col = wn * 32 + nt * 16 + c;
#pragma unroll
    for (int rr = 0; rr < 8; ++rr)
      Sf[(wm * 16 + half * 8 + rr) * 64 + col] = sacc[nt][rr];
  }
  __syncthreads();

  if (t < 64) {                        // row softmax
    float mx = -1e30f;
#pragma unroll 8
    for (int s = 0; s < 64; ++s) mx = fmaxf(mx, Sf[t * 64 + s]);
    float sum = 0.f;
#pragma unroll 8
    for (int s = 0; s < 64; ++s) {
      float e = __expf(Sf[t * 64 + s] - mx);
      sum += e;
      Ps[t * 64 + s] = (__bf16)e;
    }
    lsum[t] = sum;
  }
  __syncthreads();

  // ---- O = P . V : 64x64, K-dim = 64 ----
  v8f oacc[2] = {};
#pragma unroll
  for (int ks = 0; ks < 64; ks += 32) {
    v16bf a  = load_a_frag(Ps, 64, wm * 16, ks, lane);
    v16bf b0 = load_b_frag(Vt, 64, wn * 32, ks, lane);
    v16bf b1 = load_b_frag(Vt, 64, wn * 32 + 16, ks, lane);
    oacc[0] = WMMA_BF16(a, b0, oacc[0]);
    oacc[1] = WMMA_BF16(a, b1, oacc[1]);
  }
  __bf16* ob = attn + (size_t)(bb * 4096 + nb * 64) * 1024 + hh * 64;
#pragma unroll
  for (int nt = 0; nt < 2; ++nt) {
    int dcol = wn * 32 + nt * 16 + c;
#pragma unroll
    for (int rr = 0; rr < 8; ++rr) {
      int m = wm * 16 + half * 8 + rr;
      ob[m * 1024 + dcol] = (__bf16)(oacc[nt][rr] / (1e-6f + lsum[m]));
    }
  }
}

// ---------------------------------------------------------------------------
// Kernel 3: output projection. attn[8192,1024] bf16 x Wproj[1024,1024] f32 +
// bproj -> out f32. A tile staged by the Tensor Data Mover: each of the 8
// waves DMAs a 16-row x 64-elem bf16 slab (D# per ISA sec.8: count=1, type=2,
// data_size=2B, tile_dim0=64, tile_dim1=16, stride=1024) then waits TENSORcnt.
// ---------------------------------------------------------------------------
__global__ __launch_bounds__(256) void proj_gemm_kernel(
    const __bf16* __restrict__ A, const float* __restrict__ Wproj,
    const float* __restrict__ bproj, float* __restrict__ out) {
  __shared__ __bf16 As[128 * 64];   // [r][k] — TDM destination (linear 16KB)
  __shared__ __bf16 Bs[128 * 64];   // [n][k]

  const int t = threadIdx.x, lane = t & 31, wave = t >> 5;
  const int wm = wave & 1, wn = wave >> 1;
  const int m0 = blockIdx.y * 128, n0 = blockIdx.x * 128;

  // Wave-uniform TDM descriptor pieces (only global_addr varies with k0).
  const unsigned lds_base = (unsigned)(unsigned long long)(uintptr_t)(void*)As
                            + (unsigned)(wave * 16 * 64 * 2);
  const unsigned long long ga_base =
      (unsigned long long)(uintptr_t)A + ((unsigned long long)(m0 + wave * 16)) * 2048ull;

  v8f acc[4][2] = {};

  for (int k0 = 0; k0 < 1024; k0 += 64) {
    // ---- A tile via Tensor Data Mover (bf16 -> bf16 pure copy) ----
    {
      unsigned long long ga = ga_base + (unsigned long long)(k0 * 2);
      v4u g0;
      g0.x = 1u;                                  // count=1, user mode
      g0.y = lds_base;                            // lds_addr
      g0.z = (unsigned)ga;                        // global_addr[31:0]
      g0.w = (unsigned)(ga >> 32) | 0x80000000u;  // global_addr[56:32] | type=2
      v8u g1;
      g1.s0 = 0x10000u;        // wg_mask=0, data_size=1 (2 bytes)
      g1.s1 = 1024u << 16;     // tensor_dim0 = 1024
      g1.s2 = 8192u << 16;     // tensor_dim1 = 8192
      g1.s3 = 64u << 16;       // tile_dim0 = 64
      g1.s4 = 16u;             // tile_dim1 = 16, tile_dim2 = 0
      g1.s5 = 1024u;           // tensor_dim0_stride = 1024
      g1.s6 = 0u;
      g1.s7 = 0u;
      asm volatile("tensor_load_to_lds %0, %1" :: "s"(g0), "s"(g1) : "memory");
    }
    // ---- B tile staged manually (f32 -> bf16 conversion) ----
#pragma unroll
    for (int i = 0; i < 16; ++i) {
      int p = i * 256 + t, n = p & 127, k2 = (p >> 7) * 2;
      const float* gp = &Wproj[(k0 + k2) * 1024 + n0 + n];
      *(unsigned*)&Bs[n * 64 + k2] = pack_bf16(gp[0], gp[1024]);
    }
    __builtin_amdgcn_s_wait_tensorcnt(0);
    __syncthreads();
#pragma unroll
    for (int ks = 0; ks < 64; ks += 32) {
      v16bf a[4], b[2];
#pragma unroll
      for (int im = 0; im < 4; ++im) a[im] = load_a_frag(As, 64, wm * 64 + im * 16, ks, lane);
#pragma unroll
      for (int in_ = 0; in_ < 2; ++in_) b[in_] = load_b_frag(Bs, 64, wn * 32 + in_ * 16, ks, lane);
#pragma unroll
      for (int im = 0; im < 4; ++im)
#pragma unroll
        for (int in_ = 0; in_ < 2; ++in_)
          acc[im][in_] = WMMA_BF16(a[im], b[in_], acc[im][in_]);
    }
    __syncthreads();
  }

  const int half = lane >> 4, c = lane & 15;
#pragma unroll
  for (int im = 0; im < 4; ++im)
#pragma unroll
    for (int in_ = 0; in_ < 2; ++in_) {
      int col = n0 + wn * 32 + in_ * 16 + c;
      float bias = bproj[col];
#pragma unroll
      for (int rr = 0; rr < 8; ++rr) {
        int row = m0 + wm * 64 + im * 16 + half * 8 + rr;
        out[(size_t)row * 1024 + col] = acc[im][in_][rr] + bias;
      }
    }
}

// ---------------------------------------------------------------------------
extern "C" void kernel_launch(void* const* d_in, const int* in_sizes, int n_in,
                              void* d_out, int out_size, void* d_ws, size_t ws_size,
                              hipStream_t stream) {
  const float* x       = (const float*)d_in[0];   // [2,4096,1024]
  const float* Wqkv    = (const float*)d_in[1];   // [1024,3072]
  const float* bqkv    = (const float*)d_in[2];   // [3072]
  const float* Wproj   = (const float*)d_in[3];   // [1024,1024]
  const float* bproj   = (const float*)d_in[4];   // [1024]
  const int*   mapping = (const int*)d_in[5];     // [4096]
  float* out = (float*)d_out;                     // [2,4096,1024]

  const size_t PER = (size_t)2 * 16 * 4096 * 64;  // 8,388,608 elems per tensor
  __bf16* q    = (__bf16*)d_ws;
  __bf16* k    = q + PER;
  __bf16* v    = k + PER;
  __bf16* attn = v + PER;                         // [2,4096,1024] bf16

  qkv_gemm_kernel<<<dim3(3072 / 128, 8192 / 128), 256, 0, stream>>>(
      x, Wqkv, bqkv, q, k, v);
  hilbert_attn_kernel<<<2 * 16 * 64, 256, 0, stream>>>(q, k, v, mapping, attn);
  proj_gemm_kernel<<<dim3(1024 / 128, 8192 / 128), 256, 0, stream>>>(
      attn, Wproj, bproj, out);
}
